// OneTransBlock_50826642980982
// MI455X (gfx1250) — matrix operand order repaired and checked by reading, not compiled, not run on hardware
//
#include <hip/hip_runtime.h>
#include <math.h>

typedef __attribute__((ext_vector_type(16))) _Float16 v16h;
typedef __attribute__((ext_vector_type(8)))  _Float16 v8h;
typedef __attribute__((ext_vector_type(8)))  float    v8f;

constexpr int B_  = 256;
constexpr int S_T = 224;
constexpr int D_  = 512;
constexpr int H_  = 8;
constexpr int DH_ = 64;
constexpr int FF_ = 2048;
constexpr int SQ_ = 200;
constexpr int NS_ = 24;

enum { MAP_SEQ = 0, MAP_NS = 1, MAP_ALL = 2 };
enum { EP_QK = 0, EP_VT = 1, EP_RESID = 2, EP_GELU = 3, EP_ADDOUT = 4 };

// ---------- helpers ----------

template<int MAP>
__device__ __forceinline__ long a_row_index(int gr, int z) {
  if (MAP == MAP_SEQ) { int b = gr / SQ_; int s = gr - b * SQ_; return (long)b * S_T + s; }
  if (MAP == MAP_NS)  { return (long)gr * S_T + SQ_ + z; }
  return gr;
}

template<int MAP>
__device__ __forceinline__ void row_bs(int gr, int z, int& b, int& s) {
  if (MAP == MAP_SEQ)      { b = gr / SQ_; s = gr - b * SQ_; }
  else if (MAP == MAP_NS)  { b = gr;       s = SQ_ + z; }
  else                     { b = gr / S_T; s = gr - b * S_T; }
}

__device__ __forceinline__ v16h make_afrag(v8h lo, v8h hi) {
  return __builtin_shufflevector(lo, hi, 0,1,2,3,4,5,6,7,8,9,10,11,12,13,14,15);
}

// ---------- weight convert + transpose: W (K x N, f32) -> WT (N x K, f16) ----------

__launch_bounds__(256)
__global__ void transpose_cvt_kernel(const float* __restrict__ W,
                                     _Float16* __restrict__ WT, int K, int N) {
  __shared__ float t[32][33];
  const long mb = (long)blockIdx.z * K * N;
  const int n0 = blockIdx.x * 32, kk0 = blockIdx.y * 32;
  for (int i = threadIdx.y; i < 32; i += 8)
    t[i][threadIdx.x] = W[mb + (long)(kk0 + i) * N + n0 + threadIdx.x];
  __syncthreads();
  for (int i = threadIdx.y; i < 32; i += 8)
    WT[mb + (long)(n0 + i) * K + kk0 + threadIdx.x] = (_Float16)t[threadIdx.x][i];
}

// ---------- RMSNorm (+ optional sinusoidal positional encoding), f32 -> f16 ----------

__launch_bounds__(128)
__global__ void rmsnorm_kernel(const float* __restrict__ x,
                               const float* __restrict__ scale,
                               _Float16* __restrict__ out, int add_pe) {
  const int row = blockIdx.x;                 // b*S_T + s
  const int s = row % S_T;
  const float* xr = x + (long)row * D_;
  const int base = threadIdx.x * 4;
  float v0 = xr[base + 0], v1 = xr[base + 1], v2 = xr[base + 2], v3 = xr[base + 3];
  float ss = v0 * v0 + v1 * v1 + v2 * v2 + v3 * v3;
#pragma unroll
  for (int m = 1; m < 32; m <<= 1) ss += __shfl_xor(ss, m, 32);
  __shared__ float red[4];
  if ((threadIdx.x & 31) == 0) red[threadIdx.x >> 5] = ss;
  __syncthreads();
  const float inv = rsqrtf((red[0] + red[1] + red[2] + red[3]) * (1.0f / D_) + 1e-8f);
  float vv[4] = { v0, v1, v2, v3 };
#pragma unroll
  for (int j = 0; j < 4; j++) {
    const int d = base + j;
    float hval = vv[j] * inv * scale[d];
    if (add_pe) {
      const float div = __expf((float)(d & ~1) * (-9.210340371976184f / (float)D_));
      const float ang = (float)s * div;
      hval += (d & 1) ? __cosf(ang) : __sinf(ang);
    }
    out[(long)row * D_ + d] = (_Float16)hval;
  }
}

// ---------- generic WMMA GEMM: C = A(MxK,f16) * WT^T (WT is NxK,f16) + bias ----------
// 256 threads = 8 waves; block tile 256x64; K-step 32; wave w owns rows w*32..w*32+31
// (two 16-row groups), giving 8 WMMAs per wave per K-step against one staged W tile.

template<int MAP, int EP>
__launch_bounds__(256)
__global__ void gemm_wmma(const _Float16* __restrict__ A, int lda,
                          const _Float16* __restrict__ WT, long wzstride,
                          const float* __restrict__ bias, int bzstride,
                          const float* __restrict__ xres,
                          void* __restrict__ outp, int ldo,
                          int M, int N, int K) {
  (void)M; (void)N;
  __shared__ _Float16 As[256 * 40];
  __shared__ _Float16 Ws[64 * 40];
  const int tid  = threadIdx.x;
  const int lane = tid & 31;
  const int wave = tid >> 5;
  const int z    = blockIdx.z;
  const _Float16* W = WT + (long)z * wzstride;
  const int row0 = blockIdx.y * 256;
  const int col0 = blockIdx.x * 64;

  // cooperative tile-load indexing: thread moves 4 A chunks + 1 W chunk (16B each)
  const int ldr = tid >> 2;          // 0..63
  const int ldc = (tid & 3) * 8;     // 0,8,16,24 (halves)
  const _Float16* ap[4];
#pragma unroll
  for (int j = 0; j < 4; j++)
    ap[j] = A + a_row_index<MAP>(row0 + ldr + 64 * j, z) * (long)lda + ldc;
  const _Float16* wp = W + (long)(col0 + ldr) * K + ldc;

  v8f acc[2][4] = {};
  const int mbase = wave * 32 + (lane & 15);
  const int k0 = (lane >> 4) * 8;     // A-frag half-chunk base
  const int bk = (lane >> 4) * 16;    // unused; B frags also load at k0

  for (int kt = 0; kt < K; kt += 32) {
    v8h av[4];
#pragma unroll
    for (int j = 0; j < 4; j++) av[j] = *(const v8h*)(ap[j] + kt);
    v8h wv = *(const v8h*)(wp + kt);
    __builtin_prefetch(ap[0] + kt + 64, 0, 0);
    __builtin_prefetch(wp + kt + 64, 0, 0);
    __syncthreads();
#pragma unroll
    for (int j = 0; j < 4; j++)
      *(v8h*)&As[(ldr + 64 * j) * 40 + ldc] = av[j];
    *(v8h*)&Ws[ldr * 40 + ldc] = wv;
    __syncthreads();
    v16h b[4];
#pragma unroll
    for (int nb = 0; nb < 4; nb++)
      b[nb] = make_afrag(*(const v8h*)&Ws[(nb * 16 + (lane & 15)) * 40 + k0],
                         *(const v8h*)&Ws[(nb * 16 + (lane & 15)) * 40 + k0 + 16]);
#pragma unroll
    for (int rg = 0; rg < 2; rg++) {
      const int m = mbase + rg * 16;
      v16h a = make_afrag(*(const v8h*)&As[m * 40 + k0],
                          *(const v8h*)&As[m * 40 + k0 + 16]);
#pragma unroll
      for (int nb = 0; nb < 4; nb++)
        acc[rg][nb] = __builtin_amdgcn_wmma_f32_16x16x32_f16(
            false, a, false, b[nb], (short)0, acc[rg][nb], false, false);
    }
  }

  // epilogue: element (row0 + wave*32 + rg*16 + i + 8*(lane>>4), col0 + nb*16 + lane&15)
#pragma unroll
  for (int rg = 0; rg < 2; rg++) {
#pragma unroll
    for (int nb = 0; nb < 4; nb++) {
      const int gc = col0 + nb * 16 + (lane & 15);
      const float bvv = bias ? bias[(long)z * bzstride + gc] : 0.0f;
#pragma unroll
      for (int i = 0; i < 8; i++) {
        const int gr = row0 + wave * 32 + rg * 16 + i + 8 * (lane >> 4);
        const float v = acc[rg][nb][i] + bvv;
        if (EP == EP_QK || EP == EP_VT) {
          int bb, ssi; row_bs<MAP>(gr, z, bb, ssi);
          const int hh = gc >> 6, dd = gc & 63;
          _Float16* op = (_Float16*)outp;
          if (EP == EP_QK)
            op[(((long)bb * H_ + hh) * S_T + ssi) * DH_ + dd] = (_Float16)v;
          else
            op[(((long)bb * H_ + hh) * DH_ + dd) * S_T + ssi] = (_Float16)v;
        } else if (EP == EP_RESID) {
          const long idx = a_row_index<MAP>(gr, z) * D_ + gc;
          ((float*)outp)[idx] = xres[idx] + v;
        } else if (EP == EP_GELU) {
          const float g = 0.5f * v * (1.0f + erff(v * 0.7071067811865475f));
          ((_Float16*)outp)[a_row_index<MAP>(gr, z) * (long)ldo + gc] = (_Float16)g;
        } else { // EP_ADDOUT
          const long idx = a_row_index<MAP>(gr, z) * D_ + gc;
          ((float*)outp)[idx] += v;
        }
      }
    }
  }
}

// ---------- flash attention: 1 wave per (b, h, 16-row q tile) ----------
// q,k: (B,H,S,DH) f16 ; vt: (B,H,DH,S) f16 ; ao: (B,S,D) f16

__launch_bounds__(32)
__global__ void attention_kernel(const _Float16* __restrict__ q,
                                 const _Float16* __restrict__ k,
                                 const _Float16* __restrict__ vt,
                                 _Float16* __restrict__ ao) {
  const int lane = threadIdx.x;
  const int qt = blockIdx.x;       // 0..13
  const int h  = blockIdx.y;
  const int b  = blockIdx.z;
  __shared__ _Float16 P[16 * 40];

  const _Float16* qb = q  + (((long)b * H_ + h) * S_T) * DH_;
  const _Float16* kb = k  + (((long)b * H_ + h) * S_T) * DH_;
  const _Float16* vb = vt + (((long)b * H_ + h) * DH_) * S_T;

  const int mr = lane & 15;
  const int hi = lane >> 4;
  const int k0 = hi * 8;
  const int row_base = qt * 16;

  const _Float16* qrow = qb + (long)(row_base + mr) * DH_;
  v16h a0 = make_afrag(*(const v8h*)(qrow + k0),      *(const v8h*)(qrow + k0 + 16));
  v16h a1 = make_afrag(*(const v8h*)(qrow + 32 + k0), *(const v8h*)(qrow + 32 + k0 + 16));

  float mrow[8], lrow[8];
#pragma unroll
  for (int i = 0; i < 8; i++) { mrow[i] = -1e30f; lrow[i] = 0.0f; }
  v8f o[4] = {};

  const int jmax = (row_base + 15) / 32;        // inclusive KV-block bound (causal)
  for (int jj = 0; jj <= jmax; jj++) {
    v8f st[2];
#pragma unroll
    for (int sub = 0; sub < 2; sub++) {
      const int c0 = jj * 32 + sub * 16;
      const _Float16* krow = kb + (long)(c0 + mr) * DH_;
      v16h b0 = make_afrag(*(const v8h*)(krow + k0),      *(const v8h*)(krow + k0 + 16));
      v16h b1 = make_afrag(*(const v8h*)(krow + 32 + k0), *(const v8h*)(krow + 32 + k0 + 16));
      v8f c = {};
      c = __builtin_amdgcn_wmma_f32_16x16x32_f16(false, a0, false, b0, (short)0, c, false, false);
      c = __builtin_amdgcn_wmma_f32_16x16x32_f16(false, a1, false, b1, (short)0, c, false, false);
      const int col = c0 + mr;
#pragma unroll
      for (int i = 0; i < 8; i++) {
        const int row = row_base + i + 8 * hi;
        const float sv = c[i] * 0.125f;          // 1/sqrt(64)
        c[i] = (col > row) ? -1e30f : sv;
      }
      st[sub] = c;
    }
    // streaming softmax update (row r lives in VGPR r%8 across a 16-lane group)
#pragma unroll
    for (int i = 0; i < 8; i++) {
      float t = fmaxf(st[0][i], st[1][i]);
#pragma unroll
      for (int x = 1; x < 16; x <<= 1) t = fmaxf(t, __shfl_xor(t, x, 32));
      const float mn = fmaxf(mrow[i], t);
      const float alpha = __expf(mrow[i] - mn);
      mrow[i] = mn;
      const float p0 = __expf(st[0][i] - mn);
      const float p1 = __expf(st[1][i] - mn);
      float rs = p0 + p1;
#pragma unroll
      for (int x = 1; x < 16; x <<= 1) rs += __shfl_xor(rs, x, 32);
      lrow[i] = lrow[i] * alpha + rs;
#pragma unroll
      for (int nb = 0; nb < 4; nb++) o[nb][i] *= alpha;
      const int prow = i + 8 * hi;
      P[prow * 40 + mr]      = (_Float16)p0;
      P[prow * 40 + 16 + mr] = (_Float16)p1;
    }
    __syncthreads();
    v16h pa = make_afrag(*(const v8h*)&P[mr * 40 + k0],
                         *(const v8h*)&P[mr * 40 + k0 + 16]);
#pragma unroll
    for (int nb = 0; nb < 4; nb++) {
      const _Float16* vrow = vb + (long)(nb * 16 + mr) * S_T + jj * 32;
      v16h bv = make_afrag(*(const v8h*)(vrow + k0), *(const v8h*)(vrow + k0 + 16));
      o[nb] = __builtin_amdgcn_wmma_f32_16x16x32_f16(false, pa, false, bv, (short)0, o[nb], false, false);
    }
    __syncthreads();
  }

  _Float16* aob = ao + (long)b * S_T * D_;
#pragma unroll
  for (int nb = 0; nb < 4; nb++) {
    const int dcol = h * DH_ + nb * 16 + mr;
#pragma unroll
    for (int i = 0; i < 8; i++) {
      const int s = row_base + i + 8 * hi;
      aob[(long)s * D_ + dcol] = (_Float16)(o[nb][i] / lrow[i]);
    }
  }
}

// ---------- host orchestration ----------

extern "C" void kernel_launch(void* const* d_in, const int* in_sizes, int n_in,
                              void* d_out, int out_size, void* d_ws, size_t ws_size,
                              hipStream_t stream) {
  (void)in_sizes; (void)n_in; (void)out_size; (void)ws_size;
  const float* x      = (const float*)d_in[0];
  const float* Wq_seq = (const float*)d_in[1];
  const float* bq_seq = (const float*)d_in[2];
  const float* Wk_seq = (const float*)d_in[3];
  const float* bk_seq = (const float*)d_in[4];
  const float* Wv_seq = (const float*)d_in[5];
  const float* bv_seq = (const float*)d_in[6];
  const float* Wq_ns  = (const float*)d_in[7];
  const float* bq_ns  = (const float*)d_in[8];
  const float* Wk_ns  = (const float*)d_in[9];
  const float* bk_ns  = (const float*)d_in[10];
  const float* Wv_ns  = (const float*)d_in[11];
  const float* bv_ns  = (const float*)d_in[12];
  const float* Wo     = (const float*)d_in[13];
  const float* bo     = (const float*)d_in[14];
  const float* W1_seq = (const float*)d_in[15];
  const float* b1_seq = (const float*)d_in[16];
  const float* W2_seq = (const float*)d_in[17];
  const float* b2_seq = (const float*)d_in[18];
  const float* W1_ns  = (const float*)d_in[19];
  const float* b1_ns  = (const float*)d_in[20];
  const float* W2_ns  = (const float*)d_in[21];
  const float* b2_ns  = (const float*)d_in[22];
  const float* scale1 = (const float*)d_in[23];
  const float* scale2 = (const float*)d_in[24];

  char* wsb = (char*)d_ws;
  size_t off = 0;
  auto take = [&](size_t bytes) -> void* {
    void* p = wsb + off;
    off = (off + bytes + 255) & ~(size_t)255;
    return p;
  };
  _Float16* wq_t   = (_Float16*)take((size_t)D_ * D_ * 2);
  _Float16* wk_t   = (_Float16*)take((size_t)D_ * D_ * 2);
  _Float16* wv_t   = (_Float16*)take((size_t)D_ * D_ * 2);
  _Float16* wo_t   = (_Float16*)take((size_t)D_ * D_ * 2);
  _Float16* w1_t   = (_Float16*)take((size_t)D_ * FF_ * 2);
  _Float16* w2_t   = (_Float16*)take((size_t)FF_ * D_ * 2);
  _Float16* wqns_t = (_Float16*)take((size_t)NS_ * D_ * D_ * 2);
  _Float16* wkns_t = (_Float16*)take((size_t)NS_ * D_ * D_ * 2);
  _Float16* wvns_t = (_Float16*)take((size_t)NS_ * D_ * D_ * 2);
  _Float16* w1ns_t = (_Float16*)take((size_t)NS_ * D_ * FF_ * 2);
  _Float16* w2ns_t = (_Float16*)take((size_t)NS_ * FF_ * D_ * 2);
  _Float16* h16    = (_Float16*)take((size_t)B_ * S_T * D_ * 2);
  _Float16* q16    = (_Float16*)take((size_t)B_ * S_T * D_ * 2);
  _Float16* k16    = (_Float16*)take((size_t)B_ * S_T * D_ * 2);
  _Float16* vt16   = (_Float16*)take((size_t)B_ * S_T * D_ * 2);
  _Float16* ao16   = (_Float16*)take((size_t)B_ * S_T * D_ * 2);
  _Float16* hid16  = (_Float16*)take((size_t)B_ * S_T * FF_ * 2);

  const dim3 tb(32, 8);
  transpose_cvt_kernel<<<dim3(16, 16, 1),  tb, 0, stream>>>(Wq_seq, wq_t, 512, 512);
  transpose_cvt_kernel<<<dim3(16, 16, 1),  tb, 0, stream>>>(Wk_seq, wk_t, 512, 512);
  transpose_cvt_kernel<<<dim3(16, 16, 1),  tb, 0, stream>>>(Wv_seq, wv_t, 512, 512);
  transpose_cvt_kernel<<<dim3(16, 16, 1),  tb, 0, stream>>>(Wo,     wo_t, 512, 512);
  transpose_cvt_kernel<<<dim3(64, 16, 1),  tb, 0, stream>>>(W1_seq, w1_t, 512, 2048);
  transpose_cvt_kernel<<<dim3(16, 64, 1),  tb, 0, stream>>>(W2_seq, w2_t, 2048, 512);
  transpose_cvt_kernel<<<dim3(16, 16, NS_), tb, 0, stream>>>(Wq_ns, wqns_t, 512, 512);
  transpose_cvt_kernel<<<dim3(16, 16, NS_), tb, 0, stream>>>(Wk_ns, wkns_t, 512, 512);
  transpose_cvt_kernel<<<dim3(16, 16, NS_), tb, 0, stream>>>(Wv_ns, wvns_t, 512, 512);
  transpose_cvt_kernel<<<dim3(64, 16, NS_), tb, 0, stream>>>(W1_ns, w1ns_t, 512, 2048);
  transpose_cvt_kernel<<<dim3(16, 64, NS_), tb, 0, stream>>>(W2_ns, w2ns_t, 2048, 512);

  // h = rmsnorm(x, scale1) + PE   (f16)
  rmsnorm_kernel<<<B_ * S_T, 128, 0, stream>>>(x, scale1, h16, 1);

  // QKV seq (M = B*SQ = 51200, tile rows 256)
  gemm_wmma<MAP_SEQ, EP_QK><<<dim3(8, 200, 1), 256, 0, stream>>>(
      h16, D_, wq_t, 0, bq_seq, 0, nullptr, q16, 0, 51200, 512, 512);
  gemm_wmma<MAP_SEQ, EP_QK><<<dim3(8, 200, 1), 256, 0, stream>>>(
      h16, D_, wk_t, 0, bk_seq, 0, nullptr, k16, 0, 51200, 512, 512);
  gemm_wmma<MAP_SEQ, EP_VT><<<dim3(8, 200, 1), 256, 0, stream>>>(
      h16, D_, wv_t, 0, bv_seq, 0, nullptr, vt16, 0, 51200, 512, 512);
  // QKV ns (per-position weights, M = B = 256, z = position)
  gemm_wmma<MAP_NS, EP_QK><<<dim3(8, 1, NS_), 256, 0, stream>>>(
      h16, D_, wqns_t, (long)D_ * D_, bq_ns, D_, nullptr, q16, 0, 256, 512, 512);
  gemm_wmma<MAP_NS, EP_QK><<<dim3(8, 1, NS_), 256, 0, stream>>>(
      h16, D_, wkns_t, (long)D_ * D_, bk_ns, D_, nullptr, k16, 0, 256, 512, 512);
  gemm_wmma<MAP_NS, EP_VT><<<dim3(8, 1, NS_), 256, 0, stream>>>(
      h16, D_, wvns_t, (long)D_ * D_, bv_ns, D_, nullptr, vt16, 0, 256, 512, 512);

  // causal flash attention
  attention_kernel<<<dim3(S_T / 16, H_, B_), 32, 0, stream>>>(q16, k16, vt16, ao16);

  // x_mid = x + ao @ Wo + bo   (f32, into d_out)
  gemm_wmma<MAP_ALL, EP_RESID><<<dim3(8, 224, 1), 256, 0, stream>>>(
      ao16, D_, wo_t, 0, bo, 0, x, d_out, 0, B_ * S_T, 512, 512);

  // h2 = rmsnorm(x_mid, scale2)  (f16, reuse h buffer)
  rmsnorm_kernel<<<B_ * S_T, 128, 0, stream>>>((const float*)d_out, scale2, h16, 0);

  // FFN layer 1 (gelu), seq + ns
  gemm_wmma<MAP_SEQ, EP_GELU><<<dim3(32, 200, 1), 256, 0, stream>>>(
      h16, D_, w1_t, 0, b1_seq, 0, nullptr, hid16, FF_, 51200, 2048, 512);
  gemm_wmma<MAP_NS, EP_GELU><<<dim3(32, 1, NS_), 256, 0, stream>>>(
      h16, D_, w1ns_t, (long)FF_ * D_, b1_ns, FF_, nullptr, hid16, FF_, 256, 2048, 512);

  // FFN layer 2, accumulate onto x_mid in d_out
  gemm_wmma<MAP_SEQ, EP_ADDOUT><<<dim3(8, 200, 1), 256, 0, stream>>>(
      hid16, FF_, w2_t, 0, b2_seq, 0, nullptr, d_out, 0, 51200, 512, 2048);
  gemm_wmma<MAP_NS, EP_ADDOUT><<<dim3(8, 1, NS_), 256, 0, stream>>>(
      hid16, FF_, w2ns_t, (long)D_ * FF_, b2_ns, D_, nullptr, d_out, 0, 256, 512, 2048);
}
